// Block_25074019074700
// MI455X (gfx1250) — matrix operand order repaired and checked by reading, not compiled, not run on hardware
//
#include <hip/hip_runtime.h>

// ---------------------------------------------------------------------------
// Types for CDNA5 WMMA (gfx1250, wave32)
// ---------------------------------------------------------------------------
typedef __attribute__((ext_vector_type(16))) __bf16 v16bf;
typedef __attribute__((ext_vector_type(8)))  float  v8f;

union ABfrag { v16bf v; unsigned u[8]; unsigned short h[16]; };
union CFrag  { v8f  v; float f[8]; };
union F4     { float4 v; float f[4]; };
union U4     { uint4 v; unsigned u[4]; unsigned short h[8]; };

static constexpr int Bn  = 4;
static constexpr int Sn  = 1024;
static constexpr int En  = 1024;
static constexpr int Hn  = 16;
static constexpr int HDn = 64;
static constexpr int NEn = 8;
static constexpr int FFn = 4096;
static constexpr int BSn = Bn * Sn;          // 4096 token rows

__device__ __forceinline__ unsigned short f2bf(float f) {
    union { float f; unsigned u; } v; v.f = f;
    unsigned u = v.u;
    unsigned r = u + 0x7FFFu + ((u >> 16) & 1u);
    return (unsigned short)(r >> 16);
}
__device__ __forceinline__ unsigned pkbf(float lo, float hi) {
    return (unsigned)f2bf(lo) | ((unsigned)f2bf(hi) << 16);
}
__device__ __forceinline__ float bf2f(unsigned short h) {
    union { unsigned u; float f; } v; v.u = ((unsigned)h) << 16;
    return v.f;
}

// ---------------------------------------------------------------------------
// Generic tiled WMMA GEMM:  C[M,N] = A[M,K] * B[K,N]  (row-major, bf16 compute,
// f32 accumulate).  Macro tile 128x128, K-tile 32; 8 waves, wave tile 64x32.
// Double-buffered LDS; packed bf16-pair LDS layouts (all-b32 DS traffic).
//   AM/BM : 0 = fp32 source (convert to bf16 while staging), 1 = bf16 source
//   GA    : gather A rows through `rows[]`, M limited by *cnt
//   EPI   : 0 f32 store | 1 bf16 store | 2 relu+bf16 store |
//           3 f32 store of (acc+bias+resid) | 4 scatter: C[rows[m]] += gate*(acc+bias)
// ---------------------------------------------------------------------------
template<int AM, int BM, bool GA, int EPI>
__global__ __launch_bounds__(256)
void gemm_wmma(const void* __restrict__ Ap, const void* __restrict__ Bp,
               const float* __restrict__ bias, const float* __restrict__ resid,
               void* __restrict__ Cp,
               const int* __restrict__ rows, const int* __restrict__ cnt,
               const float* __restrict__ gate, int gstride,
               int M, int N, int K, int lda, int ldb, int ldc)
{
    // A: [128 rows][16 uints]  (uint j = bf16 pair K=2j,2j+1 of that row)
    // B: [16 k-pairs][128 cols] (uint = (B[2kp][n], B[2kp+1][n]))
    __shared__ __align__(16) unsigned As[2][128 * 16];
    __shared__ __align__(16) unsigned Bs[2][16 * 128];

    const int tid = threadIdx.x;
    const int m0  = blockIdx.y * 128;
    const int n0  = blockIdx.x * 128;

    int Mlim = M;
    if constexpr (GA || EPI == 4) { Mlim = *cnt; if (Mlim > M) Mlim = M; }
    if (m0 >= Mlim) return;

    const int wid  = tid >> 5;
    const int lane = tid & 31;
    const int wm   = wid & 1;        // 2 wave rows
    const int wn   = wid >> 1;       // 4 wave cols
    const int l15  = lane & 15;
    const int lh   = lane >> 4;      // lane half (0/1)

    CFrag acc[4][2];
#pragma unroll
    for (int i = 0; i < 4; ++i)
#pragma unroll
        for (int j = 0; j < 2; ++j)
#pragma unroll
            for (int r = 0; r < 8; ++r) acc[i][j].f[r] = 0.f;

    // A staging: thread -> (row = tid>>1, 8 uints at (tid&1)*8)
    const int ar  = tid >> 1;
    const int ac2 = (tid & 1) * 8;               // uint column offset
    const int arow_g  = m0 + ar;
    const bool avalid = arow_g < Mlim;
    long a_row;
    if constexpr (GA) a_row = avalid ? (long)rows[arow_g] : 0;
    else              a_row = avalid ? (long)arow_g : 0;

    // B staging: thread -> (k-pair = tid>>4, 8 cols at (tid&15)*8)
    const int kp = tid >> 4;
    const int nc = (tid & 15) * 8;

    auto loadA = [&](int kk, int buf) {
        unsigned* dst = &As[buf][ar * 16 + ac2];
        if constexpr (AM == 0) {
            const float* A = (const float*)Ap + (size_t)a_row * lda + kk + ac2 * 2;
            if (avalid) {
                F4 q0, q1, q2, q3;
                q0.v = ((const float4*)A)[0]; q1.v = ((const float4*)A)[1];
                q2.v = ((const float4*)A)[2]; q3.v = ((const float4*)A)[3];
                dst[0] = pkbf(q0.f[0], q0.f[1]); dst[1] = pkbf(q0.f[2], q0.f[3]);
                dst[2] = pkbf(q1.f[0], q1.f[1]); dst[3] = pkbf(q1.f[2], q1.f[3]);
                dst[4] = pkbf(q2.f[0], q2.f[1]); dst[5] = pkbf(q2.f[2], q2.f[3]);
                dst[6] = pkbf(q3.f[0], q3.f[1]); dst[7] = pkbf(q3.f[2], q3.f[3]);
            } else {
#pragma unroll
                for (int j = 0; j < 8; ++j) dst[j] = 0u;
            }
        } else {
            const unsigned short* A = (const unsigned short*)Ap + (size_t)a_row * lda + kk + ac2 * 2;
            if (avalid) {
                U4 q0, q1;
                q0.v = ((const uint4*)A)[0]; q1.v = ((const uint4*)A)[1];
#pragma unroll
                for (int j = 0; j < 4; ++j) { dst[j] = q0.u[j]; dst[4 + j] = q1.u[j]; }
            } else {
#pragma unroll
                for (int j = 0; j < 8; ++j) dst[j] = 0u;
            }
        }
    };
    auto loadB = [&](int kk, int buf) {
        unsigned* dst = &Bs[buf][kp * 128 + nc];
        if constexpr (BM == 0) {
            const float* B0 = (const float*)Bp + (size_t)(kk + 2 * kp) * ldb + n0 + nc;
            const float* B1 = B0 + ldb;
            F4 a0, a1, b0, b1;
            a0.v = ((const float4*)B0)[0]; a1.v = ((const float4*)B0)[1];
            b0.v = ((const float4*)B1)[0]; b1.v = ((const float4*)B1)[1];
#pragma unroll
            for (int j = 0; j < 4; ++j) {
                dst[j]     = pkbf(a0.f[j], b0.f[j]);
                dst[4 + j] = pkbf(a1.f[j], b1.f[j]);
            }
        } else {
            const unsigned short* B0 = (const unsigned short*)Bp + (size_t)(kk + 2 * kp) * ldb + n0 + nc;
            const unsigned short* B1 = B0 + ldb;
            U4 a0, b0;
            a0.v = ((const uint4*)B0)[0]; b0.v = ((const uint4*)B1)[0];
#pragma unroll
            for (int j = 0; j < 8; ++j)
                dst[j] = (unsigned)a0.h[j] | ((unsigned)b0.h[j] << 16);
        }
    };

    const int nkt = K >> 5;
    loadA(0, 0); loadB(0, 0);
    __syncthreads();

    for (int t = 0; t < nkt; ++t) {
        const int cur = t & 1;
        if (t + 1 < nkt) { loadA((t + 1) << 5, 1 - cur); loadB((t + 1) << 5, 1 - cur); }

        const unsigned* Ac = &As[cur][0];
        const unsigned* Bc = &Bs[cur][0];
#pragma unroll
        for (int mf = 0; mf < 4; ++mf) {
            ABfrag a;
            const int arow = wm * 64 + mf * 16 + l15;
            const int kh   = lh * 4;                 // uint offset of lane's K half
#pragma unroll
            for (int j = 0; j < 4; ++j) {
                a.u[j]     = Ac[arow * 16 + kh + j];
                a.u[4 + j] = Ac[arow * 16 + 8 + kh + j];
            }
#pragma unroll
            for (int nf = 0; nf < 2; ++nf) {
                ABfrag b;
                const int bn = wn * 32 + nf * 16 + l15;
#pragma unroll
                for (int j = 0; j < 8; ++j)
                    b.u[j] = Bc[(lh * 4 + (j & 3) + (j >> 2) * 8) * 128 + bn];
                acc[mf][nf].v = __builtin_amdgcn_wmma_f32_16x16x32_bf16(
                    false, a.v, false, b.v, (short)0, acc[mf][nf].v, false, false);
            }
        }
        __syncthreads();
    }

    // ---- epilogue ----
#pragma unroll
    for (int mf = 0; mf < 4; ++mf) {
#pragma unroll
        for (int nf = 0; nf < 2; ++nf) {
            const int ncol = n0 + wn * 32 + nf * 16 + l15;
            const float bv = bias ? bias[ncol] : 0.f;
#pragma unroll
            for (int r = 0; r < 8; ++r) {
                const int m = m0 + wm * 64 + mf * 16 + r + lh * 8;
                if (m >= Mlim) continue;
                float val = acc[mf][nf].f[r] + bv;
                if constexpr (EPI == 0) {
                    ((float*)Cp)[(size_t)m * ldc + ncol] = val;
                } else if constexpr (EPI == 1) {
                    ((unsigned short*)Cp)[(size_t)m * ldc + ncol] = f2bf(val);
                } else if constexpr (EPI == 2) {
                    ((unsigned short*)Cp)[(size_t)m * ldc + ncol] = f2bf(val > 0.f ? val : 0.f);
                } else if constexpr (EPI == 3) {
                    ((float*)Cp)[(size_t)m * ldc + ncol] = val + resid[(size_t)m * ldc + ncol];
                } else {
                    const int tok = rows[m];
                    const float g = gate[(size_t)tok * gstride];
                    ((float*)Cp)[(size_t)tok * ldc + ncol] += g * val;
                }
            }
        }
    }
    (void)gstride;
}

// ---------------------------------------------------------------------------
// LayerNorm (biased variance, eps 1e-5), one block per row of E=1024
// ---------------------------------------------------------------------------
__global__ __launch_bounds__(256)
void ln_kernel(const float* __restrict__ x, const float* __restrict__ g,
               const float* __restrict__ b, float* __restrict__ y)
{
    __shared__ float red[256];
    const int row = blockIdx.x, tid = threadIdx.x;
    const float* xr = x + (size_t)row * En;
    float s = 0.f, ss = 0.f;
    for (int i = tid; i < En; i += 256) { float v = xr[i]; s += v; ss += v * v; }
    red[tid] = s; __syncthreads();
    for (int o = 128; o > 0; o >>= 1) { if (tid < o) red[tid] += red[tid + o]; __syncthreads(); }
    const float mean = red[0] / En; __syncthreads();
    red[tid] = ss; __syncthreads();
    for (int o = 128; o > 0; o >>= 1) { if (tid < o) red[tid] += red[tid + o]; __syncthreads(); }
    const float var = red[0] / En - mean * mean;
    const float inv = rsqrtf(var + 1e-5f);
    float* yr = y + (size_t)row * En;
    for (int i = tid; i < En; i += 256) yr[i] = (xr[i] - mean) * inv * g[i] + b[i];
}

// Wq (H,E,HD) fp32 -> Wt (E, H*HD) bf16
__global__ void wqkv_transpose(const float* __restrict__ w, unsigned short* __restrict__ wt, int n)
{
    int id = blockIdx.x * 256 + threadIdx.x; if (id >= n) return;
    int e = id >> 10, c = id & 1023, h = c >> 6, d = c & 63;
    wt[id] = f2bf(w[(((size_t)h << 10) | (unsigned)e) * 64 + d]);
}

// v (B,S,E) bf16 -> vT (B,H,HD,S) bf16
__global__ void v_transpose(const unsigned short* __restrict__ v, unsigned short* __restrict__ vt, int n)
{
    int id = blockIdx.x * 256 + threadIdx.x; if (id >= n) return;
    int b = id >> 20, rem = id & 1048575, s = rem >> 10, c = rem & 1023;
    int h = c >> 6, d = c & 63;
    vt[(((size_t)(b * Hn + h) * HDn) + d) * Sn + s] = v[id];
}

// ---------------------------------------------------------------------------
// Flash attention: grid (S/128, H, B); each wave owns 16 query rows,
// streams 32-key blocks with online softmax; scale = 1/sqrt(E) = 1/32.
// ---------------------------------------------------------------------------
__global__ __launch_bounds__(256)
void attn_kernel(const unsigned short* __restrict__ q, const unsigned short* __restrict__ k,
                 const unsigned short* __restrict__ vT, unsigned short* __restrict__ o)
{
    __shared__ __align__(16) unsigned short Pls[8 * 16 * 32];
    const int b  = blockIdx.z, h = blockIdx.y;
    const int wv = threadIdx.x >> 5, lane = threadIdx.x & 31;
    const int l15 = lane & 15, lh = lane >> 4;
    const int srow0 = blockIdx.x * 128 + wv * 16;
    const float SCALE = 0.03125f;      // 1/sqrt(1024)

    // Q A-fragments (16x64 -> two K=32 chunks), loaded directly from global
    ABfrag qa[2];
    {
        const unsigned short* qp = q + ((size_t)(b * Sn + srow0 + l15)) * En + h * HDn;
#pragma unroll
        for (int c = 0; c < 2; ++c) {
            const unsigned* p0 = (const unsigned*)(qp + c * 32 + lh * 8);
            const unsigned* p1 = (const unsigned*)(qp + c * 32 + 16 + lh * 8);
#pragma unroll
            for (int j = 0; j < 4; ++j) { qa[c].u[j] = p0[j]; qa[c].u[4 + j] = p1[j]; }
        }
    }

    CFrag oacc[4];
#pragma unroll
    for (int nf = 0; nf < 4; ++nf)
#pragma unroll
        for (int r = 0; r < 8; ++r) oacc[nf].f[r] = 0.f;
    float mrow[8], lrow[8];
#pragma unroll
    for (int r = 0; r < 8; ++r) { mrow[r] = -1e30f; lrow[r] = 0.f; }

    const int nkeys = srow0 + 16;
    const int nb = (nkeys + 31) >> 5;
    for (int t0 = 0; t0 < nb * 32; t0 += 32) {
        CFrag s[2];
#pragma unroll
        for (int hh = 0; hh < 2; ++hh)
#pragma unroll
            for (int r = 0; r < 8; ++r) s[hh].f[r] = 0.f;
        // scores = Q * K^T for 32 keys (two 16-col fragments)
#pragma unroll
        for (int hh = 0; hh < 2; ++hh) {
#pragma unroll
            for (int c = 0; c < 2; ++c) {
                ABfrag kb;
                const unsigned short* kp =
                    k + ((size_t)(b * Sn + t0 + hh * 16 + l15)) * En + h * HDn + c * 32 + lh * 8;
                const unsigned* p = (const unsigned*)kp;
#pragma unroll
                for (int j = 0; j < 8; ++j) kb.u[j] = p[j + (j >> 2) * 4];
                s[hh].v = __builtin_amdgcn_wmma_f32_16x16x32_bf16(
                    false, qa[c].v, false, kb.v, (short)0, s[hh].v, false, false);
            }
        }
        // causal mask + online softmax
#pragma unroll
        for (int r = 0; r < 8; ++r) {
            const int row = srow0 + r + lh * 8;
            const int tc0 = t0 + l15, tc1 = t0 + 16 + l15;
            float x0 = (tc0 <= row) ? s[0].f[r] * SCALE : -1e30f;
            float x1 = (tc1 <= row) ? s[1].f[r] * SCALE : -1e30f;
            float mx = fmaxf(x0, x1);
            for (int d = 1; d < 16; d <<= 1) mx = fmaxf(mx, __shfl_xor(mx, d, 32));
            const float mnew = fmaxf(mrow[r], mx);
            const float fac  = __expf(mrow[r] - mnew);
            mrow[r] = mnew;
            const float p0 = __expf(x0 - mnew), p1 = __expf(x1 - mnew);
            s[0].f[r] = p0; s[1].f[r] = p1;
            float ps = p0 + p1;
            for (int d = 1; d < 16; d <<= 1) ps += __shfl_xor(ps, d, 32);
            lrow[r] = lrow[r] * fac + ps;
            oacc[0].f[r] *= fac; oacc[1].f[r] *= fac;
            oacc[2].f[r] *= fac; oacc[3].f[r] *= fac;
        }
        // re-layout P (C-frag) -> A-frag through this wave's private LDS tile
        unsigned short* Pw = Pls + wv * 512;
#pragma unroll
        for (int hh = 0; hh < 2; ++hh)
#pragma unroll
            for (int r = 0; r < 8; ++r)
                Pw[(r + lh * 8) * 32 + hh * 16 + l15] = f2bf(s[hh].f[r]);
        ABfrag pa;
        {
            const unsigned* P32 = (const unsigned*)Pw;
            const int arow = l15, kh = lh * 4;
#pragma unroll
            for (int j = 0; j < 4; ++j) {
                pa.u[j]     = P32[arow * 16 + kh + j];
                pa.u[4 + j] = P32[arow * 16 + 8 + kh + j];
            }
        }
        // O += P * V  (V^T gives contiguous B-fragment reads)
#pragma unroll
        for (int nf = 0; nf < 4; ++nf) {
            ABfrag vb;
            const unsigned short* vp =
                vT + (((size_t)(b * Hn + h) * HDn) + nf * 16 + l15) * Sn + t0 + lh * 8;
            const unsigned* p = (const unsigned*)vp;
#pragma unroll
            for (int j = 0; j < 8; ++j) vb.u[j] = p[j + (j >> 2) * 4];
            oacc[nf].v = __builtin_amdgcn_wmma_f32_16x16x32_bf16(
                false, pa.v, false, vb.v, (short)0, oacc[nf].v, false, false);
        }
    }
    // normalize + store concat-head output (bf16, [B,S,E])
#pragma unroll
    for (int nf = 0; nf < 4; ++nf)
#pragma unroll
        for (int r = 0; r < 8; ++r) {
            const int row = srow0 + r + lh * 8;
            const float val = oacc[nf].f[r] / lrow[r];
            o[((size_t)(b * Sn + row)) * En + h * HDn + nf * 16 + l15] = f2bf(val);
        }
}

// ---------------------------------------------------------------------------
// Router tail: logits[tok,8] = r1[tok,:] @ Wr2 + br2  (one block per token)
// ---------------------------------------------------------------------------
__global__ __launch_bounds__(256)
void router_logits(const unsigned short* __restrict__ r1, const float* __restrict__ Wr2,
                   const float* __restrict__ br2, float* __restrict__ logits)
{
    __shared__ float red[256];
    const int tok = blockIdx.x, tid = threadIdx.x;
    const unsigned short* rr = r1 + (size_t)tok * FFn;
    float accv[NEn];
#pragma unroll
    for (int e = 0; e < NEn; ++e) accv[e] = 0.f;
    for (int c = tid; c < FFn; c += 256) {
        const float a = bf2f(rr[c]);
        const float* w = Wr2 + (size_t)c * NEn;
#pragma unroll
        for (int e = 0; e < NEn; ++e) accv[e] += a * w[e];
    }
    for (int e = 0; e < NEn; ++e) {
        red[tid] = accv[e]; __syncthreads();
        for (int o = 128; o > 0; o >>= 1) { if (tid < o) red[tid] += red[tid + o]; __syncthreads(); }
        if (tid == 0) logits[(size_t)tok * NEn + e] = red[0] + br2[e];
        __syncthreads();
    }
}

// top-2 + softmax gate + per-expert token lists
__global__ void router_top2(const float* __restrict__ logits, float* __restrict__ gateFull,
                            int* __restrict__ cnt, int* __restrict__ lists, int ntok)
{
    int i = blockIdx.x * 256 + threadIdx.x; if (i >= ntok) return;
    const float* l = logits + (size_t)i * NEn;
    int e0 = 0; float v0 = l[0];
    for (int e = 1; e < NEn; ++e) if (l[e] > v0) { v0 = l[e]; e0 = e; }
    int e1 = -1; float v1 = -3e38f;
    for (int e = 0; e < NEn; ++e) if (e != e0 && l[e] > v1) { v1 = l[e]; e1 = e; }
    const float z = __expf(v1 - v0);
    const float inv = 1.f / (1.f + z);
    gateFull[(size_t)i * NEn + e0] = inv;
    gateFull[(size_t)i * NEn + e1] = z * inv;
    int p0 = atomicAdd(&cnt[e0], 1); lists[e0 * ntok + p0] = i;
    int p1 = atomicAdd(&cnt[e1], 1); lists[e1 * ntok + p1] = i;
}

__global__ void prep_out(const float* __restrict__ h2, float* __restrict__ out, int n)
{ int i = blockIdx.x * 256 + threadIdx.x; if (i < n) out[i] = h2[i]; }

__global__ void prep_moe(float* __restrict__ gateFull, int* __restrict__ cnt, int n)
{
    int i = blockIdx.x * 256 + threadIdx.x;
    if (i < n) gateFull[i] = 0.f;
    if (i < NEn) cnt[i] = 0;
}

// ---------------------------------------------------------------------------
extern "C" void kernel_launch(void* const* d_in, const int* in_sizes, int n_in,
                              void* d_out, int out_size, void* d_ws, size_t ws_size,
                              hipStream_t stream)
{
    (void)in_sizes; (void)n_in; (void)out_size; (void)ws_size;
    const float* x     = (const float*)d_in[0];
    const float* ln1_g = (const float*)d_in[1];
    const float* ln1_b = (const float*)d_in[2];
    const float* ln2_g = (const float*)d_in[3];
    const float* ln2_b = (const float*)d_in[4];
    const float* Wq    = (const float*)d_in[5];
    const float* bq    = (const float*)d_in[6];
    const float* Wk    = (const float*)d_in[7];
    const float* bk    = (const float*)d_in[8];
    const float* Wv    = (const float*)d_in[9];
    const float* bv    = (const float*)d_in[10];
    const float* Wo    = (const float*)d_in[11];
    const float* bo    = (const float*)d_in[12];
    const float* We1   = (const float*)d_in[13];
    const float* be1   = (const float*)d_in[14];
    const float* We2   = (const float*)d_in[15];
    const float* be2   = (const float*)d_in[16];
    const float* Wr1   = (const float*)d_in[17];
    const float* br1   = (const float*)d_in[18];
    const float* Wr2   = (const float*)d_in[19];
    const float* br2   = (const float*)d_in[20];
    float* out = (float*)d_out;

    // ---- carve workspace ----
    char* w = (char*)d_ws;
    auto alloc = [&](size_t bytes) { char* p = w; w += (bytes + 255) & ~(size_t)255; return p; };
    float*          h1    = (float*)alloc((size_t)BSn * En * 4);
    float*          x2    = (float*)alloc((size_t)BSn * En * 4);
    float*          h2    = (float*)alloc((size_t)BSn * En * 4);
    unsigned short* qb    = (unsigned short*)alloc((size_t)BSn * En * 2);
    unsigned short* kb    = (unsigned short*)alloc((size_t)BSn * En * 2);
    unsigned short* vb    = (unsigned short*)alloc((size_t)BSn * En * 2);
    unsigned short* vtb   = (unsigned short*)alloc((size_t)BSn * En * 2);
    unsigned short* ob    = (unsigned short*)alloc((size_t)BSn * En * 2);
    unsigned short* r1    = (unsigned short*)alloc((size_t)BSn * FFn * 2);
    unsigned short* mid   = (unsigned short*)alloc((size_t)BSn * FFn * 2);
    unsigned short* wqt   = (unsigned short*)alloc((size_t)En * En * 2);
    unsigned short* wkt   = (unsigned short*)alloc((size_t)En * En * 2);
    unsigned short* wvt   = (unsigned short*)alloc((size_t)En * En * 2);
    float*          logit = (float*)alloc((size_t)BSn * NEn * 4);
    float*          gateF = (float*)alloc((size_t)BSn * NEn * 4);
    int*            lists = (int*)alloc((size_t)NEn * BSn * 4);
    int*            cnt   = (int*)alloc(256);

    const dim3 blk(256);
    const int nEE = En * En;

    // 1) LN1
    ln_kernel<<<BSn, blk, 0, stream>>>(x, ln1_g, ln1_b, h1);
    // 2) repack per-head QKV weights -> bf16 [E, H*HD]
    wqkv_transpose<<<(nEE + 255) / 256, blk, 0, stream>>>(Wq, wqt, nEE);
    wqkv_transpose<<<(nEE + 255) / 256, blk, 0, stream>>>(Wk, wkt, nEE);
    wqkv_transpose<<<(nEE + 255) / 256, blk, 0, stream>>>(Wv, wvt, nEE);
    // 3) QKV projections (bf16 out)
    gemm_wmma<0,1,false,1><<<dim3(En/128, BSn/128), blk, 0, stream>>>(
        h1, wqt, bq, nullptr, qb, nullptr, nullptr, nullptr, 0,
        BSn, En, En, En, En, En);
    gemm_wmma<0,1,false,1><<<dim3(En/128, BSn/128), blk, 0, stream>>>(
        h1, wkt, bk, nullptr, kb, nullptr, nullptr, nullptr, 0,
        BSn, En, En, En, En, En);
    gemm_wmma<0,1,false,1><<<dim3(En/128, BSn/128), blk, 0, stream>>>(
        h1, wvt, bv, nullptr, vb, nullptr, nullptr, nullptr, 0,
        BSn, En, En, En, En, En);
    // 4) V -> V^T for contiguous WMMA B-fragments
    v_transpose<<<(BSn * En + 255) / 256, blk, 0, stream>>>(vb, vtb, BSn * En);
    // 5) causal flash attention
    attn_kernel<<<dim3(Sn/128, Hn, Bn), blk, 0, stream>>>(qb, kb, vtb, ob);
    // 6) output projection + bias + residual(h1)
    gemm_wmma<1,0,false,3><<<dim3(En/128, BSn/128), blk, 0, stream>>>(
        ob, Wo, bo, h1, x2, nullptr, nullptr, nullptr, 0,
        BSn, En, En, En, En, En);
    // 7) LN2
    ln_kernel<<<BSn, blk, 0, stream>>>(x2, ln2_g, ln2_b, h2);
    // 8) out = h2 ; zero gate table + expert counters
    prep_out<<<(BSn * En + 255) / 256, blk, 0, stream>>>(h2, out, BSn * En);
    prep_moe<<<(BSn * NEn + 255) / 256, blk, 0, stream>>>(gateF, cnt, BSn * NEn);
    // 9) router hidden: relu(h2 @ Wr1 + br1) -> bf16
    gemm_wmma<0,0,false,2><<<dim3(FFn/128, BSn/128), blk, 0, stream>>>(
        h2, Wr1, br1, nullptr, r1, nullptr, nullptr, nullptr, 0,
        BSn, FFn, En, En, FFn, FFn);
    // 10) router logits + top-2 gating + per-expert token lists
    router_logits<<<BSn, blk, 0, stream>>>(r1, Wr2, br2, logit);
    router_top2<<<(BSn + 255) / 256, blk, 0, stream>>>(logit, gateF, cnt, lists, BSn);
    // 11) gathered grouped-GEMM experts (sequential launches -> race-free +=)
    for (int e = 0; e < NEn; ++e) {
        const float* we1 = We1 + (size_t)e * En * FFn;
        const float* we2 = We2 + (size_t)e * FFn * En;
        gemm_wmma<0,0,true,2><<<dim3(FFn/128, BSn/128), blk, 0, stream>>>(
            h2, we1, be1 + (size_t)e * FFn, nullptr, mid,
            lists + e * BSn, cnt + e, nullptr, 0,
            BSn, FFn, En, En, FFn, FFn);
        gemm_wmma<1,0,false,4><<<dim3(En/128, BSn/128), blk, 0, stream>>>(
            mid, we2, be2 + (size_t)e * En, nullptr, out,
            lists + e * BSn, cnt + e, gateF + e, NEn,
            BSn, En, FFn, FFn, En, En);
    }
}
